// RDTAS_43645457662370
// MI455X (gfx1250) — matrix operand order repaired and checked by reading, not compiled, not run on hardware
//
#include <hip/hip_runtime.h>

// ---------------------------------------------------------------------------
// Infini-attention style segmented scan for MI455X (gfx1250), wave32 + WMMA.
// B=2, S=4096, D=1024, H=8, DK=DV=128, SEG=512, NSEG=8.
// All GEMM-shaped work goes through v_wmma_f32_16x16x32_bf16; LDS strides are
// 16B-aligned so fragment builds lower to ds_load_b128. The GEMM A-tile uses
// GLOBAL_LOAD_ASYNC_TO_LDS_B128 (ASYNCcnt) for register-free double buffering.
// ---------------------------------------------------------------------------

typedef __attribute__((ext_vector_type(16))) __bf16 v16bf;
typedef __attribute__((ext_vector_type(8)))  float  v8f;

#define LDSTR 136   // scan LDS row stride (elems): 272B, 16B-aligned, 68 dwords
#define LDSDW 68    // LDSTR/2
#define GSTR  72    // gemm LDS row stride (elems): 144B, 16B-aligned, 36 dwords

__device__ __forceinline__ float bf2f(unsigned short h) {
  return __uint_as_float(((unsigned)h) << 16);
}
__device__ __forceinline__ unsigned short f2bf(float f) {
  unsigned u = __float_as_uint(f);
  u += 0x7FFFu + ((u >> 16) & 1u);   // round-to-nearest-even
  return (unsigned short)(u >> 16);
}
__device__ __forceinline__ unsigned packbf(float lo, float hi) {
  return (unsigned)f2bf(lo) | ((unsigned)f2bf(hi) << 16);
}
__device__ __forceinline__ float sigma_act(float x) {   // elu(x)+1
  return x > 0.f ? x + 1.f : __expf(x);
}
__device__ __forceinline__ v8f vzero8() {
  v8f z;
#pragma unroll
  for (int i = 0; i < 8; ++i) z[i] = 0.f;
  return z;
}

__device__ __forceinline__ v8f wmma_bf16(v16bf a, v16bf b, v8f c) {
  return __builtin_amdgcn_wmma_f32_16x16x32_bf16(false, a, false, b, (short)0, c,
                                                 false, false);
}

// Async DMA: 16B per lane, global -> LDS, tracked by ASYNCcnt.
__device__ __forceinline__ void async_copy_b128(unsigned lds_off,
                                                const void* gaddr) {
  asm volatile("global_load_async_to_lds_b128 %0, %1, off"
               :: "v"(lds_off), "v"((unsigned long long)gaddr) : "memory");
}
__device__ __forceinline__ void wait_asynccnt0() {
  asm volatile("s_wait_asynccnt 0x0" ::: "memory");
}

// A-matrix fragment (16x32 bf16): lane l -> row (l&15); lanes 0-15 hold
// K 0..7 / 16..23, lanes 16-31 hold K 8..15 / 24..31. Two 16B runs -> 2x b128.
__device__ __forceinline__ v16bf frag_a(const unsigned short* lds, int stride,
                                        int rowBase, int k0, int lane) {
  union { v16bf v; uint4 q[2]; } f;
  const int row = rowBase + (lane & 15);
  const int kb2 = (lane < 16) ? 0 : 4;   // dword offset of K-base
  const unsigned* dw = (const unsigned*)(lds + row * stride + k0);
  f.q[0] = *(const uint4*)(dw + kb2);
  f.q[1] = *(const uint4*)(dw + kb2 + 8);
  return f.v;
}

// Same as frag_a but applies sigma = elu+1 elementwise while building.
__device__ __forceinline__ v16bf frag_a_sigma(const unsigned short* lds, int stride,
                                              int rowBase, int k0, int lane) {
  union { v16bf v; uint4 q[2]; unsigned u[8]; } f;
  const int row = rowBase + (lane & 15);
  const int kb2 = (lane < 16) ? 0 : 4;
  const unsigned* dw = (const unsigned*)(lds + row * stride + k0);
  f.q[0] = *(const uint4*)(dw + kb2);
  f.q[1] = *(const uint4*)(dw + kb2 + 8);
#pragma unroll
  for (int p = 0; p < 8; ++p) {
    const unsigned w = f.u[p];
    const float lo = sigma_act(bf2f((unsigned short)(w & 0xFFFFu)));
    const float hi = sigma_act(bf2f((unsigned short)(w >> 16)));
    f.u[p] = packbf(lo, hi);
  }
  return f.v;
}

// B-matrix fragment (32x16 bf16): lane l -> col (l&15); lanes 0-15 hold
// K 0..7 / 16..23, lanes 16-31 hold K 8..15 / 24..31. Two 16B runs -> 2x b128.
__device__ __forceinline__ v16bf frag_b(const unsigned short* lds, int stride,
                                        int colBase, int k0, int lane) {
  union { v16bf v; uint4 q[2]; } f;
  const int col = colBase + (lane & 15);
  const int kb = (lane < 16) ? 0 : 8;
  const unsigned* dw = (const unsigned*)(lds + col * stride + k0 + kb);
  f.q[0] = *(const uint4*)(dw);
  f.q[1] = *(const uint4*)(dw + 8);
  return f.v;
}

// Chain 8 WMMAs off one A-fragment (B-fragments streamed from LDS).
__device__ __forceinline__ void wmma_row(v8f acc[8], v16bf a,
                                         const unsigned short* blds, int k0,
                                         int lane) {
#pragma unroll
  for (int nt = 0; nt < 8; ++nt)
    acc[nt] = wmma_bf16(a, frag_b(blds, LDSTR, nt * 16, k0, lane), acc[nt]);
}

// ---------------------------------------------------------------------------
// fp32 -> bf16 conversion (vectorized, n multiple of 4 — true here)
// ---------------------------------------------------------------------------
__global__ void cvt_f32_bf16_kernel(const float* __restrict__ src,
                                    unsigned short* __restrict__ dst, long n4) {
  long i = (long)blockIdx.x * blockDim.x + threadIdx.x;
  const long stride = (long)gridDim.x * blockDim.x;
  for (; i < n4; i += stride) {
    const float4 v = ((const float4*)src)[i];
    unsigned* d = (unsigned*)dst + i * 2;
    d[0] = packbf(v.x, v.y);
    d[1] = packbf(v.z, v.w);
  }
}

// ---------------------------------------------------------------------------
// Generic bf16 GEMM: C[M x N] = A[M x K] * B[K x N], row-major, f32 out.
// Block tile 128x128, K-tile 64, 512 threads (16 waves, each a 16x64 strip),
// double-buffered LDS. A-tile staged via async DMA (no VGPR round-trip),
// B-tile transposed through registers. grid = (M/128, N/128, heads).
// ---------------------------------------------------------------------------
__global__ __launch_bounds__(512, 1) void gemm_bf16_nn(
    const unsigned short* __restrict__ A, long lda,
    const unsigned short* __restrict__ Bm, long ldb,
    float* __restrict__ C, long ldc,
    int Kdim, long strideBh, long strideCh) {
  __shared__ unsigned short As[2][128 * GSTR];   // [m][k], K-tile 64
  __shared__ unsigned short Bt[2][128 * GSTR];   // [n][k]

  const int tid = threadIdx.x;
  const int lane = tid & 31, wave = tid >> 5;    // 16 waves
  const int lc = lane & 15;
  const int half8 = (lane < 16) ? 0 : 8;
  const int mw = (wave & 7) * 16;                // wave row base (0..112)
  const int nw = (wave >> 3) * 64;               // wave col base (0 or 64)
  const long m0 = (long)blockIdx.x * 128;
  const long n0 = (long)blockIdx.y * 128;
  const unsigned short* Bh = Bm + (long)blockIdx.z * strideBh;
  float* Ch = C + (long)blockIdx.z * strideCh;

  uint4 rb[2];   // staged B tile (two adjacent k-rows)

  auto async_a = [&](int k0, unsigned short* as) {   // 128 rows x 64 K
#pragma unroll
    for (int i = 0; i < 2; ++i) {
      const int lin = tid + 512 * i;            // 0..1023 chunks of 16B
      const int row = lin >> 3, seg = lin & 7;
      const unsigned lds_off =
          (unsigned)(size_t)((unsigned*)as + row * (GSTR / 2) + seg * 4);
      async_copy_b128(lds_off, A + (m0 + row) * lda + k0 + seg * 8);
    }
  };
  auto load_b = [&](int k0) {
    const int kp = tid >> 4, cseg = tid & 15;   // 32 k-pairs x 16 col-segs
    rb[0] = ((const uint4*)(Bh + (long)(k0 + 2 * kp) * ldb + n0))[cseg];
    rb[1] = ((const uint4*)(Bh + (long)(k0 + 2 * kp + 1) * ldb + n0))[cseg];
  };
  auto store_b = [&](unsigned short* bt) {
    const int kp = tid >> 4, cseg = tid & 15;
    const unsigned short* p0 = (const unsigned short*)&rb[0];
    const unsigned short* p1 = (const unsigned short*)&rb[1];
    unsigned* d = (unsigned*)bt;
#pragma unroll
    for (int j = 0; j < 8; ++j)
      d[(cseg * 8 + j) * (GSTR / 2) + kp] =
          (unsigned)p0[j] | ((unsigned)p1[j] << 16);
  };

  v8f acc[4];
#pragma unroll
  for (int nt = 0; nt < 4; ++nt) acc[nt] = vzero8();

  const int nk = Kdim / 64;
  async_a(0, As[0]);
  load_b(0);
  store_b(Bt[0]);
  wait_asynccnt0();
  for (int kt = 0; kt < nk; ++kt) {
    __syncthreads();                       // tile kt visible to all waves
    if (kt + 1 < nk) {                     // overlap next-tile traffic w/ WMMA
      async_a((kt + 1) * 64, As[(kt + 1) & 1]);
      load_b((kt + 1) * 64);
    }
    const unsigned short* as = As[kt & 1];
    const unsigned short* bt = Bt[kt & 1];
#pragma unroll
    for (int ks = 0; ks < 2; ++ks) {
      const v16bf a = frag_a(as, GSTR, mw, ks * 32, lane);
#pragma unroll
      for (int nt = 0; nt < 4; ++nt)
        acc[nt] = wmma_bf16(a, frag_b(bt, GSTR, nw + nt * 16, ks * 32, lane),
                            acc[nt]);
    }
    if (kt + 1 < nk) {
      store_b(Bt[(kt + 1) & 1]);
      wait_asynccnt0();                    // A DMA done before next barrier
    }
  }

#pragma unroll
  for (int r = 0; r < 8; ++r) {
    const long row = m0 + mw + r + half8;
#pragma unroll
    for (int nt = 0; nt < 4; ++nt)
      Ch[row * ldc + n0 + nw + nt * 16 + lc] = acc[nt][r];
  }
}

// ---------------------------------------------------------------------------
// Staging helpers for the scan kernel: 128x128 f32 global tile -> bf16 LDS.
// ---------------------------------------------------------------------------
__device__ __forceinline__ void stage_rm(unsigned short* dst, const float* src,
                                         int tid) {
#pragma unroll
  for (int i = 0; i < 16; ++i) {
    const int lin = tid + 256 * i;
    const int row = lin >> 5, c4 = lin & 31;
    const float4 v = ((const float4*)(src + (long)row * 128))[c4];
    uint2 t;
    t.x = packbf(v.x, v.y);
    t.y = packbf(v.z, v.w);
    ((uint2*)((unsigned*)dst + row * LDSDW))[c4] = t;   // b64 store
  }
}
// Transposed staging: processes row pairs so writes are packed b32.
__device__ __forceinline__ void stage_tr(unsigned short* dst, const float* src,
                                         int tid) {
#pragma unroll
  for (int i = 0; i < 8; ++i) {
    const int lin = tid + 256 * i;            // 0..2047
    const int rp = lin >> 5, c4 = lin & 31;   // 64 row-pairs x 32 col4s
    const float4 v0 = ((const float4*)(src + (long)(2 * rp) * 128))[c4];
    const float4 v1 = ((const float4*)(src + (long)(2 * rp + 1) * 128))[c4];
    unsigned* d = (unsigned*)dst;
    const int n0 = c4 * 4;
    d[(n0 + 0) * LDSDW + rp] = packbf(v0.x, v1.x);
    d[(n0 + 1) * LDSDW + rp] = packbf(v0.y, v1.y);
    d[(n0 + 2) * LDSDW + rp] = packbf(v0.z, v1.z);
    d[(n0 + 3) * LDSDW + rp] = packbf(v0.w, v1.w);
  }
}

// ---------------------------------------------------------------------------
// Segmented scan: one block per (b,h). Flash-style causal softmax attention,
// linear-attention memory read, delta-rule memory update — all via WMMA.
// K/V/Q layout: (H, B*S, 128) f32. attb: (B*S, H*128) bf16.
// ---------------------------------------------------------------------------
__global__ __launch_bounds__(256, 1) void infini_scan_kernel(
    const float* __restrict__ Kf, const float* __restrict__ Vf,
    const float* __restrict__ Qf, const float* __restrict__ betas,
    unsigned short* __restrict__ attb) {
  constexpr int S = 4096, SEG = 512, NSEG = 8, DK = 128, DV = 128, H = 8;
  constexpr long BS = 2L * S;
  const int b = blockIdx.x / H, h = blockIdx.x % H;
  const int tid = threadIdx.x;
  const int lane = tid & 31, wave = tid >> 5;
  const int lc = lane & 15;
  const int half8 = (lane < 16) ? 0 : 8;
  const int prow_t = tid & 127, phalf = tid >> 7;   // denominator partials
  const float scale = 0.08838834764831845f;         // 1/sqrt(DK)

  __shared__ float memF[DK * DV];               // fp32 master memory
  __shared__ unsigned short memB[DV * LDSTR];   // bf16 memory, [dv][dk]
  __shared__ unsigned short KsB[128 * LDSTR];   // k block [s][dk] (raw / sigma)
  __shared__ unsigned short VtB[128 * LDSTR];   // v block [dv][s]
  __shared__ unsigned short QsKt[128 * LDSTR];  // q chunk [i][dk] / sigma_kT [dk][s]
  __shared__ unsigned short PsUt[128 * LDSTR];  // P [i][j] / u [dv][s] / att out
  __shared__ float zv[128], gateS[128];
  __shared__ float den2[2][128], zn2[2][128];

  const long pbase = ((long)h * BS + (long)b * S) * DK;

  for (int i = tid; i < DK * DV; i += 256) memF[i] = 0.f;
  for (int i = tid; i < DV * LDSTR; i += 256) memB[i] = 0;
  if (tid < 128) {
    zv[tid] = 1.0f / 128.0f;
    gateS[tid] = 1.0f / (1.0f + __expf(-betas[h * DV + tid]));
  }
  __syncthreads();

  for (int seg = 0; seg < NSEG; ++seg) {
    const int s0 = seg * SEG;

    // ---------------- attention phase (uses pre-update mem, z) -------------
    for (int c = 0; c < 4; ++c) {
      const int i0 = c * 128;
      __syncthreads();
      stage_rm(QsKt, Qf + pbase + (long)(s0 + i0) * DK, tid);
      __syncthreads();
      {  // den2 partials: sigma_q(row) . z  (2 threads per row)
        float sacc = 0.f;
        const int d0 = phalf * 64;
        for (int d = d0; d < d0 + 64; ++d)
          sacc += sigma_act(bf2f(QsKt[prow_t * LDSTR + d])) * zv[d];
        den2[phalf][prow_t] = sacc;
      }

      float mrow[8], lrow[8];
      v8f o[8];
#pragma unroll
      for (int r = 0; r < 8; ++r) { mrow[r] = -3.0e38f; lrow[r] = 0.f; o[r] = vzero8(); }

      for (int jb = 0; jb <= c; ++jb) {   // causal: only j-blocks <= i-chunk
        __syncthreads();
        stage_rm(KsB, Kf + pbase + (long)(s0 + jb * 128) * DK, tid);
        stage_tr(VtB, Vf + pbase + (long)(s0 + jb * 128) * DV, tid);
        __syncthreads();

        v8f sacc[8];
#pragma unroll
        for (int nt = 0; nt < 8; ++nt) sacc[nt] = vzero8();
#pragma unroll
        for (int ks = 0; ks < 4; ++ks)
          wmma_row(sacc, frag_a(QsKt, LDSTR, wave * 16, ks * 32, lane),
                   KsB, ks * 32, lane);

        const bool diag = (jb == c);
#pragma unroll
        for (int r = 0; r < 8; ++r) {
          const int iloc = i0 + wave * 16 + r + half8;
          float pv[8];
          float mx = -3.0e38f;
#pragma unroll
          for (int nt = 0; nt < 8; ++nt) {
            float val = sacc[nt][r] * scale;
            if (diag && (jb * 128 + nt * 16 + lc) > iloc) val = -3.0e38f;
            pv[nt] = val;
            mx = fmaxf(mx, val);
          }
#pragma unroll
          for (int m = 1; m <= 8; m <<= 1) mx = fmaxf(mx, __shfl_xor(mx, m, 32));
          const float mnew = fmaxf(mrow[r], mx);
          const float corr = __expf(mrow[r] - mnew);
          float rs = 0.f;
          const int prow = wave * 16 + r + half8;
#pragma unroll
          for (int nt = 0; nt < 8; ++nt) {
            const float p = __expf(pv[nt] - mnew);
            rs += p;
            PsUt[prow * LDSTR + nt * 16 + lc] = f2bf(p);
            o[nt][r] *= corr;
          }
#pragma unroll
          for (int m = 1; m <= 8; m <<= 1) rs += __shfl_xor(rs, m, 32);
          lrow[r] = lrow[r] * corr + rs;
          mrow[r] = mnew;
        }

#pragma unroll
        for (int ks = 0; ks < 4; ++ks)   // O += P @ V
          wmma_row(o, frag_a(PsUt, LDSTR, wave * 16, ks * 32, lane),
                   VtB, ks * 32, lane);
      }

      // linear-attention read: macc = sigma_q @ mem
      v8f macc[8];
#pragma unroll
      for (int nt = 0; nt < 8; ++nt) macc[nt] = vzero8();
#pragma unroll
      for (int ks = 0; ks < 4; ++ks)
        wmma_row(macc, frag_a_sigma(QsKt, LDSTR, wave * 16, ks * 32, lane),
                 memB, ks * 32, lane);

      // gated combine into PsUt (bf16 [row][dv]), then coalesced store
#pragma unroll
      for (int r = 0; r < 8; ++r) {
        const int rloc = wave * 16 + r + half8;
        const float dn = den2[0][rloc] + den2[1][rloc];
        const float li = lrow[r];
#pragma unroll
        for (int nt = 0; nt < 8; ++nt) {
          const int dv = nt * 16 + lc;
          const float g = gateS[dv];
          const float val = g * (macc[nt][r] / dn) + (1.f - g) * (o[nt][r] / li);
          PsUt[rloc * LDSTR + dv] = f2bf(val);
        }
      }
      __syncthreads();
#pragma unroll
      for (int i = 0; i < 4; ++i) {
        const int lin = tid + 256 * i;            // 0..1023
        const int row = lin >> 3, seg = lin & 7;  // 8 x 16B: elems 0..63 of each row
        const uint4 t = *(const uint4*)((const unsigned*)PsUt + row * LDSDW + seg * 4);
        *(uint4*)(attb + ((long)b * S + s0 + i0 + row) * 1024 + h * DV + seg * 8) = t;
      }
    }

    // ---------------- delta-rule memory update ----------------------------
    __syncthreads();
    zn2[phalf][prow_t] = 0.f;
    v8f dacc[8];
#pragma unroll
    for (int nt = 0; nt < 8; ++nt) dacc[nt] = vzero8();

    for (int c = 0; c < 4; ++c) {
      __syncthreads();
      // stage sigma_k row-major (KsB) and transposed pair-packed (QsKt as Kt)
#pragma unroll
      for (int i = 0; i < 8; ++i) {
        const int lin = tid + 256 * i;
        const int rp = lin >> 5, c4 = lin & 31;
        const float* srck = Kf + pbase + (long)(s0 + c * 128) * DK;
        const float4 v0 = ((const float4*)(srck + (long)(2 * rp) * 128))[c4];
        const float4 v1 = ((const float4*)(srck + (long)(2 * rp + 1) * 128))[c4];
        const float a0 = sigma_act(v0.x), a1 = sigma_act(v0.y);
        const float a2 = sigma_act(v0.z), a3 = sigma_act(v0.w);
        const float b0 = sigma_act(v1.x), b1 = sigma_act(v1.y);
        const float b2 = sigma_act(v1.z), b3 = sigma_act(v1.w);
        uint2 t0; t0.x = packbf(a0, a1); t0.y = packbf(a2, a3);
        uint2 t1; t1.x = packbf(b0, b1); t1.y = packbf(b2, b3);
        ((uint2*)((unsigned*)KsB + (2 * rp) * LDSDW))[c4] = t0;
        ((uint2*)((unsigned*)KsB + (2 * rp + 1) * LDSDW))[c4] = t1;
        unsigned* d = (unsigned*)QsKt;
        const int dk0 = c4 * 4;
        d[(dk0 + 0) * LDSDW + rp] = packbf(a0, b0);
        d[(dk0 + 1) * LDSDW + rp] = packbf(a1, b1);
        d[(dk0 + 2) * LDSDW + rp] = packbf(a2, b2);
        d[(dk0 + 3) * LDSDW + rp] = packbf(a3, b3);
      }
      __syncthreads();
      {  // den2[s] partial: sigma_k(s).z_old; zn2[dk] partial: column sums
        float sm = 0.f, cs = 0.f;
        const int d0 = phalf * 64;
        for (int d = d0; d < d0 + 64; ++d)
          sm += bf2f(KsB[prow_t * LDSTR + d]) * zv[d];
        for (int s = d0; s < d0 + 64; ++s)
          cs += bf2f(KsB[s * LDSTR + prow_t]);
        den2[phalf][prow_t] = sm;
        zn2[phalf][prow_t] += cs;
      }
      __syncthreads();

      // corr = sigma_k @ mem
      v8f cacc[8];
#pragma unroll
      for (int nt = 0; nt < 8; ++nt) cacc[nt] = vzero8();
#pragma unroll
      for (int ks = 0; ks < 4; ++ks)
        wmma_row(cacc, frag_a(KsB, LDSTR, wave * 16, ks * 32, lane),
                 memB, ks * 32, lane);

      // u = v - corr/denom, stored transposed in PsUt[dv][s]
#pragma unroll
      for (int r = 0; r < 8; ++r) {
        const int sloc = wave * 16 + r + half8;
        const float dn = den2[0][sloc] + den2[1][sloc];
        const long sg = (long)(s0 + c * 128 + sloc);
#pragma unroll
        for (int nt = 0; nt < 8; ++nt) {
          const int dv = nt * 16 + lc;
          const float vvv = Vf[pbase + sg * DV + dv];
          PsUt[dv * LDSTR + sloc] = f2bf(vvv - cacc[nt][r] / dn);
        }
      }
      __syncthreads();

      // dacc += sigma_k^T @ u   (rank update, rows = dk)
#pragma unroll
      for (int ks = 0; ks < 4; ++ks)
        wmma_row(dacc, frag_a(QsKt, LDSTR, wave * 16, ks * 32, lane),
                 PsUt, ks * 32, lane);
    }

    __syncthreads();
#pragma unroll
    for (int r = 0; r < 8; ++r) {
      const int dk = wave * 16 + r + half8;
#pragma unroll
      for (int nt = 0; nt < 8; ++nt) {
        const int dv = nt * 16 + lc;
        const float f = memF[dk * DV + dv] + dacc[nt][r];
        memF[dk * DV + dv] = f;
        memB[dv * LDSTR + dk] = f2bf(f);
      }
    }
    if (tid < 128) zv[tid] += zn2[0][tid] + zn2[1][tid];
    __syncthreads();
  }
}

// ---------------------------------------------------------------------------
// Launcher
// ---------------------------------------------------------------------------
extern "C" void kernel_launch(void* const* d_in, const int* in_sizes, int n_in,
                              void* d_out, int out_size, void* d_ws, size_t ws_size,
                              hipStream_t stream) {
  (void)in_sizes; (void)n_in; (void)out_size; (void)ws_size;
  const float* x     = (const float*)d_in[0];
  const float* Wk    = (const float*)d_in[1];
  const float* Wv    = (const float*)d_in[2];
  const float* Wq    = (const float*)d_in[3];
  const float* Wout  = (const float*)d_in[4];
  const float* betas = (const float*)d_in[5];
  float* out = (float*)d_out;

  const long BS = 8192, D = 1024, DK = 128, H = 8;

  char* ws = (char*)d_ws;
  size_t off = 0;
  auto carve = [&](size_t bytes) -> char* {
    char* p = ws + off;
    off += (bytes + 255) & ~(size_t)255;
    return p;
  };
  unsigned short* xb    = (unsigned short*)carve(BS * D * 2);          // 16 MB
  unsigned short* Wkb   = (unsigned short*)carve(H * D * DK * 2);      //  2 MB
  unsigned short* Wvb   = (unsigned short*)carve(H * D * DK * 2);
  unsigned short* Wqb   = (unsigned short*)carve(H * D * DK * 2);
  unsigned short* Woutb = (unsigned short*)carve(H * DK * D * 2);
  float* Kf = (float*)carve(H * BS * DK * 4);                          // 32 MB
  float* Vf = (float*)carve(H * BS * DK * 4);
  float* Qf = (float*)carve(H * BS * DK * 4);
  unsigned short* attb = (unsigned short*)carve(BS * D * 2);           // 16 MB

  // fp32 -> bf16
  cvt_f32_bf16_kernel<<<2048, 256, 0, stream>>>(x, xb, (BS * D) / 4);
  cvt_f32_bf16_kernel<<<512, 256, 0, stream>>>(Wk, Wkb, (H * D * DK) / 4);
  cvt_f32_bf16_kernel<<<512, 256, 0, stream>>>(Wv, Wvb, (H * D * DK) / 4);
  cvt_f32_bf16_kernel<<<512, 256, 0, stream>>>(Wq, Wqb, (H * D * DK) / 4);
  cvt_f32_bf16_kernel<<<512, 256, 0, stream>>>(Wout, Woutb, (H * DK * D) / 4);

  // projections: (BS x D) @ (D x DK) per head -> (H, BS, DK) f32
  dim3 gp(64, 1, 8);
  gemm_bf16_nn<<<gp, 512, 0, stream>>>(xb, D, Wkb, DK, Kf, DK, (int)D,
                                       D * DK, BS * DK);
  gemm_bf16_nn<<<gp, 512, 0, stream>>>(xb, D, Wvb, DK, Vf, DK, (int)D,
                                       D * DK, BS * DK);
  gemm_bf16_nn<<<gp, 512, 0, stream>>>(xb, D, Wqb, DK, Qf, DK, (int)D,
                                       D * DK, BS * DK);

  // segmented scan (serial over NSEG per (b,h))
  infini_scan_kernel<<<16, 256, 0, stream>>>(Kf, Vf, Qf, betas, attb);

  // out = attb (BS x 1024) @ Wout (1024 x 1024), f32
  gemm_bf16_nn<<<dim3(64, 8, 1), 512, 0, stream>>>(attb, D, Woutb, D, out, D,
                                                   (int)D, 0L, 0L);
}
